// ACDE_87531433492502
// MI455X (gfx1250) — matrix operand order repaired and checked by reading, not compiled, not run on hardware
//
#include <hip/hip_runtime.h>
#include <math.h>

// ---------------------------------------------------------------------------
// Problem constants (match reference)
// ---------------------------------------------------------------------------
#define NPIX 65536
#define FDIM 224
#define PEND 6
#define HDIM 256

// GEMM tiling
#define ROWS_PER_WG 64     // 4 waves x 16 rows
#define WG_THREADS 128
#define ACT_STRIDE 264     // bf16 elements; row byte-stride 528 = 33*16 (16B aligned)

typedef __attribute__((ext_vector_type(16))) __bf16 v16bf;
typedef __attribute__((ext_vector_type(8)))  float  v8f;

// natural alignment of v16bf is 32 bytes (>= the 16B our uint4 views need)
union BFrag { v16bf v; unsigned short s[16]; uint4 q[2]; };

// float -> bf16 (round to nearest even), as raw u16
__device__ __forceinline__ unsigned short f2bf(float f) {
    unsigned u = __float_as_uint(f);
    unsigned r = u + 0x7fffu + ((u >> 16) & 1u);
    return (unsigned short)(r >> 16);
}

// order-preserving float<->uint encoding so we can use integer atomicMax
__device__ __forceinline__ unsigned enc_f32(float f) {
    unsigned u = __float_as_uint(f);
    return (u & 0x80000000u) ? ~u : (u | 0x80000000u);
}
__device__ __forceinline__ float dec_f32(unsigned e) {
    unsigned v = (e & 0x80000000u) ? (e & 0x7fffffffu) : ~e;
    return __uint_as_float(v);
}

// Async 16B global->LDS copy (ASYNCcnt-tracked, no VGPR round trip).
// lds_off = LDS byte offset (low 32 bits of generic shared address),
// voff    = per-lane byte offset, sbase = uniform 64-bit base.
__device__ __forceinline__ void async_copy_b128(unsigned lds_off, unsigned voff,
                                                const void* sbase) {
    asm volatile("global_load_async_to_lds_b128 %0, %1, %2"
                 :: "v"(lds_off), "v"(voff), "s"(sbase) : "memory");
}
__device__ __forceinline__ void wait_async0() {
    asm volatile("s_wait_asynccnt 0x0" ::: "memory");
}

// ---------------------------------------------------------------------------
// K0: per-pixel argmax over the 6 abundances (first-max, like jnp.argmax),
//     plus zero-init of the global reduction buffers.
// ---------------------------------------------------------------------------
__global__ void k_classify(const float* __restrict__ ab, int* __restrict__ cls,
                           unsigned* __restrict__ colmax_u,
                           float* __restrict__ numer, float* __restrict__ denom) {
    int n = blockIdx.x * blockDim.x + threadIdx.x;
    if (n < NPIX) {
        float best = ab[(size_t)n * PEND];
        int bi = 0;
        #pragma unroll
        for (int p = 1; p < PEND; ++p) {
            float v = ab[(size_t)n * PEND + p];
            if (v > best) { best = v; bi = p; }
        }
        cls[n] = bi;
    }
    if (n < PEND * FDIM) {
        colmax_u[n] = 0u;       // enc of smallest key
        numer[n] = 0.0f;
        denom[n] = 0.0f;
    }
}

// ---------------------------------------------------------------------------
// K-pre: one-time fp32 -> bf16 conversion of a weight matrix into the
// fragment-blocked layout consumed directly by the WMMA B operand:
//   Wpk[(kslice*NT + t)*512 + lane*16 + i] = bf16(W[k0 + kmap(lane,i)][t*16+col(lane)])
// ---------------------------------------------------------------------------
template <int KDIM, int NCOLS>
__global__ void __launch_bounds__(256)
k_prepack(const float* __restrict__ W, unsigned short* __restrict__ Wpk) {
    constexpr int NT = NCOLS / 16;
    constexpr int NGROUP = (KDIM / 32) * NT * 64;   // 8-element groups
    int gg = blockIdx.x * blockDim.x + threadIdx.x;
    if (gg >= NGROUP) return;
    int kslice = gg / (NT * 64);
    int g      = gg % (NT * 64);
    int t      = g >> 6;
    int r      = g & 63;
    int lane_d = r >> 1;
    int i8     = r & 1;
    int kk = ((lane_d >> 4) << 3) + i8 * 16;
    int c  = t * 16 + (lane_d & 15);
    const float* src = W + (size_t)(kslice * 32 + kk) * NCOLS + c;
    unsigned short tmp[8];
    #pragma unroll
    for (int j = 0; j < 8; ++j) tmp[j] = f2bf(src[(size_t)j * NCOLS]);
    uint4 pk;
    pk.x = (unsigned)tmp[0] | ((unsigned)tmp[1] << 16);
    pk.y = (unsigned)tmp[2] | ((unsigned)tmp[3] << 16);
    pk.z = (unsigned)tmp[4] | ((unsigned)tmp[5] << 16);
    pk.w = (unsigned)tmp[6] | ((unsigned)tmp[7] << 16);
    *(uint4*)(Wpk + (size_t)gg * 8) = pk;
}

// ---------------------------------------------------------------------------
// K1: fused 3-layer MLP with bf16 WMMA (f32 accumulate).
//     Prepacked bf16 weights are async-copied (global -> LDS) 16KB per K step;
//     A fragments are two ds_load_b128; B fragments are double-buffered
//     ds_load_b128 pairs so WMMA overlaps the next tile's loads.
// ---------------------------------------------------------------------------
template <int KDIM, int NCOLS>
__device__ __forceinline__ void layer_gemm(const unsigned short* __restrict__ Wpk,
                                           unsigned short* sWf,
                                           const unsigned short* sActWave,
                                           v8f* acc, int tid, int lane) {
    constexpr int NT = NCOLS / 16;
    v8f zero = {};
    #pragma unroll
    for (int t = 0; t < NT; ++t) acc[t] = zero;

    const int ar   = lane & 15;        // A row (M)
    const int koff = (lane >> 4) * 8;  // A K sub-offset per lane half

    for (int k0 = 0; k0 < KDIM; k0 += 32) {
        __syncthreads();   // all waves done reading previous sWf slice
        // Async-copy this K-slice's prepacked fragments: NT*64 x 16B chunks.
        const unsigned short* src = Wpk + (size_t)(k0 >> 5) * NT * 512;
        for (int g = tid; g < NT * 64; g += WG_THREADS) {
            unsigned lds_off = (unsigned)(unsigned long long)(sWf + g * 8);
            async_copy_b128(lds_off, (unsigned)(g * 16), src);
        }
        wait_async0();
        __syncthreads();

        // A fragment: row `ar`, K = k0 + {koff..koff+7, 16+koff..16+koff+7}
        BFrag A;
        const unsigned short* ap = sActWave + ar * ACT_STRIDE + k0 + koff;
        A.q[0] = *(const uint4*)(ap);
        A.q[1] = *(const uint4*)(ap + 16);

        // Double-buffered B fragments
        BFrag Bf[2];
        {
            const unsigned short* bp = sWf + lane * 16;
            Bf[0].q[0] = *(const uint4*)(bp);
            Bf[0].q[1] = *(const uint4*)(bp + 8);
        }
        #pragma unroll
        for (int t = 0; t < NT; ++t) {
            if (t + 1 < NT) {
                const unsigned short* bp = sWf + (t + 1) * 512 + lane * 16;
                Bf[(t + 1) & 1].q[0] = *(const uint4*)(bp);
                Bf[(t + 1) & 1].q[1] = *(const uint4*)(bp + 8);
            }
            acc[t] = __builtin_amdgcn_wmma_f32_16x16x32_bf16(
                false, A.v, false, Bf[t & 1].v, (short)0, acc[t], false, false);
        }
    }
}

template <int NCOLS>
__device__ __forceinline__ void act_store_relu(const v8f* acc,
                                               const float* __restrict__ bias,
                                               unsigned short* sActWave, int lane) {
    const int mbase = (lane >> 4) * 8;  // C/D: VGPR j -> row j (+8 for hi half)
    const int c0 = lane & 15;
    #pragma unroll
    for (int t = 0; t < NCOLS / 16; ++t) {
        int c = t * 16 + c0;
        float b = bias[c];
        #pragma unroll
        for (int j = 0; j < 8; ++j) {
            float v = acc[t][j] + b;
            v = v > 0.0f ? v : 0.0f;
            sActWave[(mbase + j) * ACT_STRIDE + c] = f2bf(v);
        }
    }
}

__global__ void __launch_bounds__(WG_THREADS)
k_mlp(const float* __restrict__ Y,
      const unsigned short* __restrict__ Wp1, const float* __restrict__ b1,
      const unsigned short* __restrict__ Wp2, const float* __restrict__ b2,
      const unsigned short* __restrict__ Wp3, const float* __restrict__ b3,
      const int* __restrict__ cls,
      float* __restrict__ logits, unsigned* __restrict__ colmax_u) {
    __shared__ __align__(16) unsigned short sAct[ROWS_PER_WG * ACT_STRIDE]; // 33 KB
    __shared__ __align__(16) unsigned short sWf[16 * 512];                  // 16 KB
    __shared__ unsigned sMax[PEND * FDIM];                                  // 5.25 KB

    const int tid  = threadIdx.x;
    const int lane = tid & 31;
    const int wave = tid >> 5;
    const int row0 = blockIdx.x * ROWS_PER_WG;

    for (int i = tid; i < PEND * FDIM; i += WG_THREADS) sMax[i] = 0u;

    // stage Y tile (64 x 224) as bf16 pairs (float2 in, b32 out)
    for (int i = tid; i < ROWS_PER_WG * (FDIM / 2); i += WG_THREADS) {
        int r = i / (FDIM / 2), c2 = i % (FDIM / 2);
        float2 y = *(const float2*)(Y + (size_t)(row0 + r) * FDIM + c2 * 2);
        unsigned v = (unsigned)f2bf(y.x) | ((unsigned)f2bf(y.y) << 16);
        *(unsigned*)(sAct + r * ACT_STRIDE + c2 * 2) = v;
    }
    __syncthreads();

    unsigned short* sActWave = &sAct[wave * 16 * ACT_STRIDE];
    v8f acc[16];

    layer_gemm<FDIM, HDIM>(Wp1, sWf, sActWave, acc, tid, lane);   // 224 -> 256
    act_store_relu<HDIM>(acc, b1, sActWave, lane);

    layer_gemm<HDIM, HDIM>(Wp2, sWf, sActWave, acc, tid, lane);   // 256 -> 256
    act_store_relu<HDIM>(acc, b2, sActWave, lane);

    layer_gemm<HDIM, FDIM>(Wp3, sWf, sActWave, acc, tid, lane);   // 256 -> 224

    // logits: +b3, store fp32, and accumulate per-(class,col) max in LDS
    {
        const int mbase = (lane >> 4) * 8;
        const int c0 = lane & 15;
        const int grow = row0 + wave * 16;
        #pragma unroll
        for (int t = 0; t < FDIM / 16; ++t) {
            int c = t * 16 + c0;
            float b = b3[c];
            #pragma unroll
            for (int j = 0; j < 8; ++j) {
                int n = grow + mbase + j;
                float v = acc[t][j] + b;
                logits[(size_t)n * FDIM + c] = v;
                int p = cls[n];
                atomicMax(&sMax[p * FDIM + c], enc_f32(v));
            }
        }
    }
    __syncthreads();
    for (int i = tid; i < PEND * FDIM; i += WG_THREADS) {
        unsigned m = sMax[i];
        if (m) atomicMax(&colmax_u[i], m);
    }
}

// ---------------------------------------------------------------------------
// K2: numer[p,f] += exp(l - max)*Y ; denom[p,f] += exp(l - max)
//     One thread per feature column -> conflict-free LDS partials.
// ---------------------------------------------------------------------------
__global__ void __launch_bounds__(FDIM)
k_expsum(const float* __restrict__ Y, const float* __restrict__ logits,
         const int* __restrict__ cls, const unsigned* __restrict__ colmax_u,
         float* __restrict__ numer, float* __restrict__ denom, int rows_per_block) {
    __shared__ float s_num[PEND * FDIM];
    __shared__ float s_den[PEND * FDIM];
    const int f = threadIdx.x;  // 0..223, owns column f for every class
    for (int i = f; i < PEND * FDIM; i += FDIM) { s_num[i] = 0.0f; s_den[i] = 0.0f; }

    float cmax[PEND];
    #pragma unroll
    for (int p = 0; p < PEND; ++p) cmax[p] = dec_f32(colmax_u[p * FDIM + f]);

    int n0 = blockIdx.x * rows_per_block;
    int n1 = n0 + rows_per_block; if (n1 > NPIX) n1 = NPIX;
    for (int n = n0; n < n1; ++n) {
        int p = cls[n];
        float l = logits[(size_t)n * FDIM + f];
        float e = expf(l - cmax[p]);
        s_num[p * FDIM + f] += e * Y[(size_t)n * FDIM + f];
        s_den[p * FDIM + f] += e;
    }
    #pragma unroll
    for (int p = 0; p < PEND; ++p) {
        atomicAdd(&numer[p * FDIM + f], s_num[p * FDIM + f]);
        atomicAdd(&denom[p * FDIM + f], s_den[p * FDIM + f]);
    }
}

// ---------------------------------------------------------------------------
// K3: M = numer / denom, zero for empty classes (denom==0 <=> count==0)
// ---------------------------------------------------------------------------
__global__ void k_finalM(const float* __restrict__ numer,
                         const float* __restrict__ denom, float* __restrict__ M) {
    int i = blockIdx.x * blockDim.x + threadIdx.x;
    if (i < PEND * FDIM) {
        float d = denom[i];
        M[i] = (d > 0.0f) ? numer[i] / d : 0.0f;
    }
}

// ---------------------------------------------------------------------------
// K4: Y_hat = abundance @ M   (bandwidth bound; M cached in LDS)
// ---------------------------------------------------------------------------
__global__ void __launch_bounds__(FDIM)
k_yhat(const float* __restrict__ ab, const float* __restrict__ M,
       float* __restrict__ out, int rows_per_block) {
    __shared__ float sM[PEND * FDIM];
    const int f = threadIdx.x;
    for (int i = f; i < PEND * FDIM; i += FDIM) sM[i] = M[i];
    __syncthreads();

    int n0 = blockIdx.x * rows_per_block;
    int n1 = n0 + rows_per_block; if (n1 > NPIX) n1 = NPIX;
    for (int n = n0; n < n1; ++n) {
        float a0 = ab[(size_t)n * PEND + 0];
        float a1 = ab[(size_t)n * PEND + 1];
        float a2 = ab[(size_t)n * PEND + 2];
        float a3 = ab[(size_t)n * PEND + 3];
        float a4 = ab[(size_t)n * PEND + 4];
        float a5 = ab[(size_t)n * PEND + 5];
        float s = a0 * sM[0 * FDIM + f] + a1 * sM[1 * FDIM + f] +
                  a2 * sM[2 * FDIM + f] + a3 * sM[3 * FDIM + f] +
                  a4 * sM[4 * FDIM + f] + a5 * sM[5 * FDIM + f];
        out[(size_t)n * FDIM + f] = s;
    }
}

// ---------------------------------------------------------------------------
// Host-side launch
// ---------------------------------------------------------------------------
extern "C" void kernel_launch(void* const* d_in, const int* in_sizes, int n_in,
                              void* d_out, int out_size, void* d_ws, size_t ws_size,
                              hipStream_t stream) {
    const float* ab = (const float*)d_in[0];
    const float* Y  = (const float*)d_in[1];
    const float* W1 = (const float*)d_in[2];
    const float* b1 = (const float*)d_in[3];
    const float* W2 = (const float*)d_in[4];
    const float* b2 = (const float*)d_in[5];
    const float* W3 = (const float*)d_in[6];
    const float* b3 = (const float*)d_in[7];
    float* out = (float*)d_out;

    char* ws = (char*)d_ws;
    float*    logits   = (float*)ws;    ws += (size_t)NPIX * FDIM * sizeof(float);
    int*      cls      = (int*)ws;      ws += (size_t)NPIX * sizeof(int);
    unsigned* colmax_u = (unsigned*)ws; ws += (size_t)PEND * FDIM * sizeof(unsigned);
    float*    numer    = (float*)ws;    ws += (size_t)PEND * FDIM * sizeof(float);
    float*    denom    = (float*)ws;    ws += (size_t)PEND * FDIM * sizeof(float);
    float*    M        = (float*)ws;    ws += (size_t)PEND * FDIM * sizeof(float);
    unsigned short* Wp1 = (unsigned short*)ws; ws += (size_t)FDIM * HDIM * 2;
    unsigned short* Wp2 = (unsigned short*)ws; ws += (size_t)HDIM * HDIM * 2;
    unsigned short* Wp3 = (unsigned short*)ws; ws += (size_t)HDIM * FDIM * 2;
    (void)ws_size; (void)in_sizes; (void)n_in; (void)out_size;

    k_classify<<<NPIX / 256, 256, 0, stream>>>(ab, cls, colmax_u, numer, denom);
    k_prepack<FDIM, HDIM><<<(FDIM * HDIM / 8 + 255) / 256, 256, 0, stream>>>(W1, Wp1);
    k_prepack<HDIM, HDIM><<<(HDIM * HDIM / 8 + 255) / 256, 256, 0, stream>>>(W2, Wp2);
    k_prepack<HDIM, FDIM><<<(HDIM * FDIM / 8 + 255) / 256, 256, 0, stream>>>(W3, Wp3);
    k_mlp<<<NPIX / ROWS_PER_WG, WG_THREADS, 0, stream>>>(Y, Wp1, b1, Wp2, b2, Wp3, b3,
                                                         cls, logits, colmax_u);
    k_expsum<<<128, FDIM, 0, stream>>>(Y, logits, cls, colmax_u, numer, denom,
                                       NPIX / 128);
    k_finalM<<<(PEND * FDIM + 255) / 256, 256, 0, stream>>>(numer, denom, M);
    k_yhat<<<256, FDIM, 0, stream>>>(ab, M, out, NPIX / 256);
}
